// MultiHeadStacked_56444460204441
// MI455X (gfx1250) — matrix operand order repaired and checked by reading, not compiled, not run on hardware
//
#include <hip/hip_runtime.h>
#include <stdint.h>

#define Bdim 2
#define Sdim 2048
#define Ddim 1024
#define Hdim 16
#define Edim 64

typedef __bf16 bf16_t;
typedef __attribute__((ext_vector_type(16))) __bf16   v16bf;
typedef __attribute__((ext_vector_type(8)))  float    v8f;
typedef __attribute__((ext_vector_type(4)))  uint32_t uvec4;

union FragB16 { v16bf v; uvec4 u[2]; };

__device__ __forceinline__ bf16_t f2bf(float f) {
  union { float f; uint32_t u; } in; in.f = f;
  uint32_t u = in.u;
  uint32_t r = u + 0x7FFFu + ((u >> 16) & 1u);       // round-to-nearest-even
  union { uint16_t s; bf16_t b; } out; out.s = (uint16_t)(r >> 16);
  return out.b;
}

// Async global->LDS copy of one 16-byte chunk per lane (CDNA5 ASYNCcnt path).
__device__ __forceinline__ void async_copy_b128(uint32_t lds_byte_off, const bf16_t* g) {
  asm volatile("global_load_async_to_lds_b128 %0, %1, off"
               :: "v"(lds_byte_off), "v"(g) : "memory");
}

// ---------------------------------------------------------------------------
// Kernel 1: fp32 -> bf16; x kept [B][S][D]; W written TRANSPOSED [3][H][E][D]
// (so the async copy engine can stage GEMM B-tiles without a VGPR transpose).
// Wq pre-scaled by 1/sqrt(E). Destination-linear indexing -> coalesced stores.
// ---------------------------------------------------------------------------
__global__ void cvt_kernel(const float* __restrict__ x,  const float* __restrict__ wq,
                           const float* __restrict__ wk, const float* __restrict__ wv,
                           bf16_t* __restrict__ xb, bf16_t* __restrict__ wb) {
  const int XN = Bdim * Sdim * Ddim;
  const int WN = Hdim * Ddim * Edim;
  int i = blockIdx.x * 256 + threadIdx.x;
  int total = XN + 3 * WN;
  if (i >= total) return;
  if (i < XN) { xb[i] = f2bf(x[i]); return; }
  int j = i - XN;                       // linear over destination [3][H][E][D]
  int which = j / WN, rem = j - which * WN;
  int h  = rem / (Edim * Ddim);  rem -= h * (Edim * Ddim);
  int n  = rem / Ddim;
  int k  = rem - n * Ddim;
  const float* W = (which == 0) ? wq : (which == 1 ? wk : wv);
  float v = W[(size_t)h * Ddim * Edim + (size_t)k * Edim + n];
  if (which == 0) v *= 0.125f;          // fold 1/sqrt(E) into Wq
  wb[j] = f2bf(v);
}

// ---------------------------------------------------------------------------
// Kernel 2: QKV projection GEMM, bf16 WMMA, double-buffered async W staging.
//   block = 8 waves, 128 M-rows; wave = 16 rows x 64 cols (4 f32 accumulators).
//   W tile (64 n x 32 k) async-copied into LDS (stride 40 bf16, 16B aligned,
//   conflict-free ds_load_b128 B-fragments). Ping-pong buffers hide the copy.
//   V output stored transposed [E][S] for the PV stage.
// ---------------------------------------------------------------------------
__global__ void qkv_gemm_kernel(const bf16_t* __restrict__ xb, const bf16_t* __restrict__ wb,
                                bf16_t* __restrict__ qb, bf16_t* __restrict__ kbuf,
                                bf16_t* __restrict__ vtb) {
  __shared__ __align__(16) bf16_t ldsW[2][64 * 40];
  const int bh    = blockIdx.y;
  const int qkv   = blockIdx.z;
  const int b     = bh / Hdim, h = bh % Hdim;
  const int mbase = blockIdx.x * 128;
  const int wave  = threadIdx.x >> 5;
  const int lane  = threadIdx.x & 31;
  const int lh    = lane >> 4;        // 0: lanes 0-15, 1: lanes 16-31
  const int ln    = lane & 15;

  const bf16_t* xrow  = xb + ((size_t)b * Sdim + mbase + wave * 16 + ln) * Ddim;
  const bf16_t* whead = wb + ((size_t)qkv * Hdim + h) * Edim * Ddim;   // [E][D]

  // this thread's 16-byte chunk of the 64x32 W tile
  const int cn = threadIdx.x >> 2;          // 0..63 : n row
  const int cc = threadIdx.x & 3;           // 0..3  : 8-elem k chunk
  const bf16_t* wsrc = whead + (size_t)cn * Ddim + cc * 8;
  const uint32_t loff0 = (uint32_t)(uintptr_t)(&ldsW[0][cn * 40 + cc * 8]);
  const uint32_t loff1 = (uint32_t)(uintptr_t)(&ldsW[1][cn * 40 + cc * 8]);

  v8f acc[4];
  #pragma unroll
  for (int t = 0; t < 4; ++t)
    #pragma unroll
    for (int i = 0; i < 8; ++i) acc[t][i] = 0.f;

  async_copy_b128(loff0, wsrc);                       // prologue: tile 0 -> buf 0

  for (int k0 = 0; k0 < Ddim; k0 += 32) {
    const int cur = (k0 >> 5) & 1;
    if (k0 + 32 < Ddim) {                             // prefetch next tile
      async_copy_b128(cur ? loff0 : loff1, wsrc + k0 + 32);
      asm volatile("s_wait_asynccnt 0x1" ::: "memory");  // tile k0 landed (in-order)
    } else {
      asm volatile("s_wait_asynccnt 0x0" ::: "memory");
    }
    __syncthreads();                                  // whole tile visible to block

    FragB16 a;                                        // A: 16x32 of x (global B128s)
    const bf16_t* ap = xrow + k0 + (lh ? 8 : 0);
    a.u[0] = *(const uvec4*)ap;
    a.u[1] = *(const uvec4*)(ap + 16);

    const bf16_t* wtile = ldsW[cur];
    #pragma unroll
    for (int t = 0; t < 4; ++t) {                     // B: 32x16 of W (LDS B128s)
      FragB16 bm;
      const bf16_t* bp = &wtile[(t * 16 + ln) * 40 + (lh ? 8 : 0)];
      bm.u[0] = *(const uvec4*)bp;
      bm.u[1] = *(const uvec4*)(bp + 16);
      acc[t] = __builtin_amdgcn_wmma_f32_16x16x32_bf16(
          false, a.v, false, bm.v, (short)0, acc[t], false, false);
    }
    __syncthreads();                                  // done reading before reuse
  }

  #pragma unroll
  for (int t = 0; t < 4; ++t) {
    #pragma unroll
    for (int r = 0; r < 8; ++r) {
      int orow = mbase + wave * 16 + r + (lh ? 8 : 0);
      int n = t * 16 + ln;
      bf16_t val = f2bf(acc[t][r]);
      if (qkv == 0)      qb  [((size_t)bh * Sdim + orow) * Edim + n] = val;
      else if (qkv == 1) kbuf[((size_t)bh * Sdim + orow) * Edim + n] = val;
      else               vtb [((size_t)bh * Edim + n) * Sdim + orow] = val;  // V^T
    }
  }
}

// ---------------------------------------------------------------------------
// Kernel 3: causal flash attention. One 16-row query tile per wave.
//   key tiles of 64: 8 WMMAs (QK^T) + online softmax + 8 WMMAs (PV).
//   P converted D-layout -> A-layout via per-wave LDS (in-order DS, wave-local).
// ---------------------------------------------------------------------------
__global__ void attn_kernel(const bf16_t* __restrict__ qb, const bf16_t* __restrict__ kbuf,
                            const bf16_t* __restrict__ vtb, float* __restrict__ out) {
  __shared__ __align__(16) bf16_t ldsP[8 * 16 * 72];
  const int bh = blockIdx.y;
  const int b  = bh / Hdim, h = bh % Hdim;
  const int wave = threadIdx.x >> 5, lane = threadIdx.x & 31;
  const int lh = lane >> 4, ln = lane & 15;
  const int qbase = (blockIdx.x * 8 + wave) * 16;

  const bf16_t* Q  = qb   + (size_t)bh * Sdim * Edim;
  const bf16_t* Kp = kbuf + (size_t)bh * Sdim * Edim;
  const bf16_t* Vt = vtb  + (size_t)bh * Edim * Sdim;
  bf16_t* Pl = &ldsP[wave * 16 * 72];

  float m[8], l[8];
  v8f o[4];
  #pragma unroll
  for (int r = 0; r < 8; ++r) { m[r] = -1e30f; l[r] = 0.f; }
  #pragma unroll
  for (int t = 0; t < 4; ++t)
    #pragma unroll
    for (int i = 0; i < 8; ++i) o[t][i] = 0.f;

  const int nfull = qbase / 64;                 // tiles < nfull are fully unmasked
  for (int kt = 0; kt <= nfull; ++kt) {
    v8f s[4];
    #pragma unroll
    for (int t = 0; t < 4; ++t)
      #pragma unroll
      for (int i = 0; i < 8; ++i) s[t][i] = 0.f;

    #pragma unroll
    for (int e0 = 0; e0 < 64; e0 += 32) {       // scores: Q (16x32) @ K^T (32x16) x4
      FragB16 a;
      const bf16_t* ap = Q + (size_t)(qbase + ln) * Edim + e0 + (lh ? 8 : 0);
      a.u[0] = *(const uvec4*)ap; a.u[1] = *(const uvec4*)(ap + 16);
      #pragma unroll
      for (int t = 0; t < 4; ++t) {
        const bf16_t* bp = Kp + (size_t)(kt * 64 + t * 16 + ln) * Edim + e0 + (lh ? 8 : 0);
        FragB16 bm; bm.u[0] = *(const uvec4*)bp; bm.u[1] = *(const uvec4*)(bp + 16);
        s[t] = __builtin_amdgcn_wmma_f32_16x16x32_bf16(
            false, a.v, false, bm.v, (short)0, s[t], false, false);
      }
    }

    if (kt == nfull) {                          // causal mask on boundary tile
      #pragma unroll
      for (int t = 0; t < 4; ++t) {
        int col = kt * 64 + t * 16 + ln;
        #pragma unroll
        for (int r = 0; r < 8; ++r) {
          int row = qbase + r + (lh ? 8 : 0);
          if (col > row) s[t][r] = -1e30f;
        }
      }
    }

    #pragma unroll
    for (int r = 0; r < 8; ++r) {               // online softmax (per-row, 16-lane halves)
      float mx = fmaxf(fmaxf(s[0][r], s[1][r]), fmaxf(s[2][r], s[3][r]));
      for (int msk = 1; msk < 16; msk <<= 1) mx = fmaxf(mx, __shfl_xor(mx, msk, 32));
      float mn = fmaxf(m[r], mx);
      float fr = __expf(m[r] - mn);
      m[r] = mn;
      float rs = 0.f;
      #pragma unroll
      for (int t = 0; t < 4; ++t) { float p = __expf(s[t][r] - mn); s[t][r] = p; rs += p; }
      for (int msk = 1; msk < 16; msk <<= 1) rs += __shfl_xor(rs, msk, 32);
      l[r] = l[r] * fr + rs;
      #pragma unroll
      for (int t = 0; t < 4; ++t) o[t][r] *= fr;
    }

    #pragma unroll
    for (int t = 0; t < 4; ++t)                 // P (D-layout) -> LDS row-major
      #pragma unroll
      for (int r = 0; r < 8; ++r)
        Pl[(r + (lh ? 8 : 0)) * 72 + t * 16 + ln] = f2bf(s[t][r]);
    __builtin_amdgcn_wave_barrier();
    asm volatile("" ::: "memory");

    #pragma unroll
    for (int kb0 = 0; kb0 < 64; kb0 += 32) {    // PV: P (16x32) @ V (32x16) x4
      FragB16 a;
      const bf16_t* ap = &Pl[ln * 72 + kb0 + (lh ? 8 : 0)];
      a.u[0] = *(const uvec4*)ap; a.u[1] = *(const uvec4*)(ap + 16);
      #pragma unroll
      for (int t = 0; t < 4; ++t) {
        const bf16_t* vp = Vt + (size_t)(t * 16 + ln) * Sdim + kt * 64 + kb0 + (lh ? 8 : 0);
        FragB16 bm; bm.u[0] = *(const uvec4*)vp; bm.u[1] = *(const uvec4*)(vp + 16);
        o[t] = __builtin_amdgcn_wmma_f32_16x16x32_bf16(
            false, a.v, false, bm.v, (short)0, o[t], false, false);
      }
    }
    __builtin_amdgcn_wave_barrier();
    asm volatile("" ::: "memory");
  }

  #pragma unroll
  for (int t = 0; t < 4; ++t) {
    #pragma unroll
    for (int r = 0; r < 8; ++r) {
      int row = qbase + r + (lh ? 8 : 0);
      int e = t * 16 + ln;
      out[(((size_t)b * Sdim + row) * Hdim + h) * Edim + e] = o[t][r] / l[r];
    }
  }
}

// ---------------------------------------------------------------------------
extern "C" void kernel_launch(void* const* d_in, const int* in_sizes, int n_in,
                              void* d_out, int out_size, void* d_ws, size_t ws_size,
                              hipStream_t stream) {
  (void)in_sizes; (void)n_in; (void)out_size; (void)ws_size;
  const float* x  = (const float*)d_in[0];
  const float* wq = (const float*)d_in[1];
  const float* wk = (const float*)d_in[2];
  const float* wv = (const float*)d_in[3];

  const size_t XN = (size_t)Bdim * Sdim * Ddim;        // x elems
  const size_t WN = (size_t)Hdim * Ddim * Edim;        // one W elems
  const size_t QN = (size_t)Bdim * Hdim * Sdim * Edim; // q/k/v elems

  bf16_t* xb   = (bf16_t*)d_ws;        // ~38 MB of workspace total
  bf16_t* wb   = xb + XN;              // transposed W [3][H][E][D]
  bf16_t* qb   = wb + 3 * WN;
  bf16_t* kbuf = qb + QN;
  bf16_t* vtb  = kbuf + QN;

  {
    size_t total = XN + 3 * WN;
    dim3 g((unsigned)((total + 255) / 256));
    cvt_kernel<<<g, 256, 0, stream>>>(x, wq, wk, wv, xb, wb);
  }
  {
    dim3 g(Sdim / 128, Bdim * Hdim, 3);
    qkv_gemm_kernel<<<g, 256, 0, stream>>>(xb, wb, qb, kbuf, vtb);
  }
  {
    dim3 g(Sdim / 128, Bdim * Hdim);
    attn_kernel<<<g, 256, 0, stream>>>(qb, kbuf, vtb, (float*)d_out);
  }
}
